// AttentionHead_52587579572821
// MI455X (gfx1250) — matrix operand order repaired and checked by reading, not compiled, not run on hardware
//
#include <hip/hip_runtime.h>
#include <hip/hip_bf16.h>

// ---------------------------------------------------------------------------
// AttentionHead (B=4, T=2048, D=1024, HS=64) for MI455X / gfx1250.
// HBM-bound on rel_pos_embd (1 GiB fp32) -> stream it exactly once, NT hints.
// Kernel 0: one-time weight convert/transpose f32[D,HS] -> bf16[HS,D].
// Kernel 1: QKV projection, software-pipelined v_wmma_f32_16x16x32_bf16.
// Kernel 2: flash attention + rel bias; one workgroup per 16-query tile,
//           all 4 batches in-block so rel is read once; K/V frag loads are
//           hoisted ahead of the rel-streaming phase to hide latency.
// ---------------------------------------------------------------------------

typedef __bf16 bf16;
typedef bf16         v16bf __attribute__((ext_vector_type(16)));
typedef float        v8f   __attribute__((ext_vector_type(8)));
typedef float        v4f   __attribute__((ext_vector_type(4)));
typedef unsigned int v4u   __attribute__((ext_vector_type(4)));

#define B_   4
#define T_   2048
#define D_   1024
#define HS_  64
#define MROWS (B_ * T_)   // 8192

#define WMMA_BF16(A, Bf, C) \
  __builtin_amdgcn_wmma_f32_16x16x32_bf16(false, (A), false, (Bf), (short)0, (C), false, false)

union FragU { v4u u[2]; v16bf v; };

__device__ __forceinline__ v8f zero8() {
  v8f z;
#pragma unroll
  for (int i = 0; i < 8; ++i) z[i] = 0.0f;
  return z;
}

// ---- A fragment (16x32 bf16) from row-major [rows x pitch] bf16 ----
// ISA layout: lanes 0-15 -> row=lane, K in {0..7,16..23};
//             lanes 16-31 -> row=lane-16, K in {8..15,24..31}.
__device__ __forceinline__ v16bf load_a_frag_bf16(const bf16* __restrict__ base,
                                                  int pitch, int lane) {
  const int row   = lane & 15;
  const int kbase = (lane >> 4) << 3;  // 0 or 8
  const bf16* p = base + (size_t)row * pitch + kbase;
  FragU f;
  f.u[0] = *(const v4u*)(p);        // K = kbase .. kbase+7
  f.u[1] = *(const v4u*)(p + 16);   // K = kbase+16 .. kbase+23
  return f.v;
}

// ---- A fragment (16x32) from row-major f32, converted to bf16 ----
__device__ __forceinline__ v16bf load_a_frag_f32(const float* __restrict__ base,
                                                 int pitch, int lane) {
  const int row   = lane & 15;
  const int kbase = (lane >> 4) << 3;
  const float* p = base + (size_t)row * pitch + kbase;
  v16bf a;
#pragma unroll
  for (int i = 0; i < 8; ++i) {
    const int k = ((i & 3) << 1) + ((i >> 2) << 4);  // 0,2,4,6,16,18,20,22
    const float2 f = *(const float2*)(p + k);
    a[2 * i]     = (bf16)f.x;
    a[2 * i + 1] = (bf16)f.y;
  }
  return a;
}

// ---- B fragment (32x16 bf16): logical B[k][n] = base[n*pitch + k]
// (K-contiguous source). Per lane: two b128 loads.
// ISA layout: lanes 0-15 -> col n=lane, K in {0..7,16..23}; lanes 16-31 -> K in {8..15,24..31}.
__device__ __forceinline__ v16bf load_b_frag_tr(const bf16* __restrict__ base,
                                                int pitch, int lane) {
  const int n  = lane & 15;
  const int kb = (lane >> 4) << 3;  // 0 or 8
  const bf16* p = base + (size_t)n * pitch + kb;
  FragU f;
  f.u[0] = *(const v4u*)(p);
  f.u[1] = *(const v4u*)(p + 16);
  return f.v;
}

// ===========================================================================
// Kernel 0: one-time weight transpose+convert: w[D][HS] f32 -> wT[HS][D] bf16
// for each of q/k/v. Tiny (768 KB total).
// ===========================================================================
__global__ void __launch_bounds__(256) ah_wt_convert(
    const float* __restrict__ wq, const float* __restrict__ wk,
    const float* __restrict__ wv, bf16* __restrict__ wT) {
  const int idx   = blockIdx.x * 256 + threadIdx.x;
  const int total = 3 * HS_ * D_;
  if (idx >= total) return;
  const int p = idx / (HS_ * D_);
  const int r = idx - p * (HS_ * D_);
  const int d = r / HS_;      // consecutive threads -> consecutive h (coalesced read)
  const int h = r - d * HS_;
  const float* w = (p == 0) ? wq : (p == 1) ? wk : wv;
  wT[((size_t)p * HS_ + h) * D_ + d] = (bf16)w[(size_t)d * HS_ + h];
}

// ===========================================================================
// Kernel 1: fused QKV projection.  grid = 512 (16-row M tiles of [B*T, D]),
// block = 96 threads = 3 waves (wave w -> q/k/v).  Each wave: 16x64 tile,
// K loop over D in steps of 32 -> 128 WMMAs, software-pipelined so the
// next step's A/B fragments load under the current step's WMMAs.
// Outputs: q,k bf16 row-major [B*T,HS]; v bf16 TRANSPOSED [B][HS][T];
//          q additionally in f32 for the VALU rel-bias path.
// ===========================================================================
__global__ void __launch_bounds__(96) ah_qkv_proj(
    const float* __restrict__ x, const bf16* __restrict__ wT,
    const float* __restrict__ bq, const float* __restrict__ bk,
    const float* __restrict__ bv,
    bf16* __restrict__ qb, bf16* __restrict__ kb, bf16* __restrict__ vT,
    float* __restrict__ qf) {
  const int lane = threadIdx.x & 31;
  const int wave = threadIdx.x >> 5;  // 0=q, 1=k, 2=v
  const int m0   = blockIdx.x * 16;

  const bf16*  wp   = wT + (size_t)wave * HS_ * D_;
  const float* bia  = (wave == 0) ? bq : (wave == 1) ? bk : bv;
  const float* xrow = x + (size_t)m0 * D_;

  v8f acc[4];
#pragma unroll
  for (int n = 0; n < 4; ++n) acc[n] = zero8();

  // --- software pipeline: prologue loads for k0 = 0 ---
  v16bf a = load_a_frag_f32(xrow, D_, lane);
  v16bf bfr[4];
#pragma unroll
  for (int n = 0; n < 4; ++n)
    bfr[n] = load_b_frag_tr(wp + (size_t)(n * 16) * D_, D_, lane);

  for (int k0 = 0; k0 < D_ - 32; k0 += 32) {
    // issue next step's loads first -> they fly under this step's WMMAs
    const int k1 = k0 + 32;
    const v16bf a_n = load_a_frag_f32(xrow + k1, D_, lane);
    v16bf b_n[4];
#pragma unroll
    for (int n = 0; n < 4; ++n)
      b_n[n] = load_b_frag_tr(wp + (size_t)(n * 16) * D_ + k1, D_, lane);

#pragma unroll
    for (int n = 0; n < 4; ++n) acc[n] = WMMA_BF16(a, bfr[n], acc[n]);

    a = a_n;
#pragma unroll
    for (int n = 0; n < 4; ++n) bfr[n] = b_n[n];
  }
  // --- epilogue: last K step ---
#pragma unroll
  for (int n = 0; n < 4; ++n) acc[n] = WMMA_BF16(a, bfr[n], acc[n]);

  // C layout: element j -> row = j + 8*(lane>=16), col = lane&15 (per n-tile).
  const int colc  = lane & 15;
  const int rbase = (lane >> 4) << 3;
#pragma unroll
  for (int n = 0; n < 4; ++n) {
#pragma unroll
    for (int j = 0; j < 8; ++j) {
      const int   r = m0 + j + rbase;          // global row in [0, B*T)
      const int   h = n * 16 + colc;
      const float v = acc[n][j] + bia[h];
      if (wave == 0) {
        qb[(size_t)r * HS_ + h] = (bf16)v;
        qf[(size_t)r * HS_ + h] = v;
      } else if (wave == 1) {
        kb[(size_t)r * HS_ + h] = (bf16)v;
      } else {
        const int bb = r >> 11, tt = r & (T_ - 1);
        vT[((size_t)bb * HS_ + h) * T_ + tt] = (bf16)v;  // transposed store
      }
    }
  }
}

// ===========================================================================
// Kernel 2: flash attention with relative-position bias.
// grid = T/16 = 128 blocks; block = 256 threads = 8 waves.
//   - all 8 waves stream the rel bias (1 GiB, read ONCE, non-temporal)
//   - waves 0..3 each own batch b=wave: WMMA scores, online softmax, P*V.
//   - K/V fragments for the current key block are loaded BEFORE the rel
//     streaming phase, so their latency is hidden behind it.
// ===========================================================================
__global__ void __launch_bounds__(256, 1) ah_attention(
    const float* __restrict__ rel,
    const bf16* __restrict__ qb, const bf16* __restrict__ kb,
    const bf16* __restrict__ vT, const float* __restrict__ qf,
    float* __restrict__ out) {
  __shared__ float qf_s[B_][16][HS_];    // 16 KB: q rows (f32) for bias dots
  __shared__ float bias_s[B_][16][32];   //  8 KB: rel bias for current 32 keys
  __shared__ bf16  p_s[B_][16][32];      //  4 KB: P tile C->A layout transpose

  const int tid  = threadIdx.x;
  const int lane = tid & 31;
  const int wave = tid >> 5;   // 0..7
  const int t0   = blockIdx.x * 16;

  // Stage q (f32) for this t-tile, all batches, into LDS.
  for (int idx = tid; idx < B_ * 16 * HS_; idx += 256) {
    const int b = idx >> 10, t = (idx >> 6) & 15, c = idx & 63;
    qf_s[b][t][c] = qf[((size_t)(b * T_ + t0 + t)) * HS_ + c];
  }
  __syncthreads();

  // Per-wave persistent flash-attention state (waves 0..3; batch b = wave).
  const int b = wave & 3;
  const bf16* kt0 = kb + (size_t)b * T_ * HS_;       // [T][HS] for batch b
  const bf16* vtb = vT + (size_t)b * HS_ * T_;       // [HS][T] for batch b
  v16bf qa0, qa1;
  float mrow[8], lrow[8];
  v8f   o[4];
  if (wave < 4) {
    const bf16* qbase = qb + ((size_t)(b * T_ + t0)) * HS_;
    qa0 = load_a_frag_bf16(qbase, HS_, lane);        // head dims 0..31
    qa1 = load_a_frag_bf16(qbase + 32, HS_, lane);   // head dims 32..63
#pragma unroll
    for (int j = 0; j < 8; ++j) { mrow[j] = -3.0e38f; lrow[j] = 0.0f; }
#pragma unroll
    for (int n = 0; n < 4; ++n) o[n] = zero8();
  }

  const int colc  = lane & 15;
  const int rbase = (lane >> 4) << 3;

  for (int it = 0; it < T_ / 32; ++it) {
    const int v0 = it * 32;

    // ---- Hoisted operand loads for this key block (waves 0..3). These
    // issue before the rel-streaming phase; loads complete in order, so the
    // WMMAs in Phase B see them already resident.
    v16bf kf[4], vf[4];
    if (wave < 4) {
      const bf16* kt = kt0 + (size_t)v0 * HS_;
      kf[0] = load_b_frag_tr(kt, HS_, lane);             // keys v0..+15, c 0..31
      kf[1] = load_b_frag_tr(kt + 32, HS_, lane);        // c 32..63
      kf[2] = load_b_frag_tr(kt + 16 * HS_, HS_, lane);  // keys +16
      kf[3] = load_b_frag_tr(kt + 16 * HS_ + 32, HS_, lane);
#pragma unroll
      for (int n = 0; n < 4; ++n)  // B[w][h'] = vT[n*16+h'][v0+w]
        vf[n] = load_b_frag_tr(vtb + (size_t)(n * 16) * T_ + v0, T_, lane);
    }

    // ---- Phase A (all 8 waves): rel bias for 16 t-rows x 32 keys x 4 batches.
    // rel has zero reuse: non-temporal b128 streams.
    for (int p = tid; p < 16 * 32; p += 256) {
      const int t = p >> 5, v = p & 31;
      const v4f* r =
          (const v4f*)(rel + ((size_t)(t0 + t) * T_ + (v0 + v)) * HS_);
      float d0 = 0.f, d1 = 0.f, d2 = 0.f, d3 = 0.f;
#pragma unroll
      for (int c4 = 0; c4 < HS_ / 4; ++c4) {
        const v4f rr = __builtin_nontemporal_load(r + c4);
        const v4f q0 = *(const v4f*)&qf_s[0][t][4 * c4];
        const v4f q1 = *(const v4f*)&qf_s[1][t][4 * c4];
        const v4f q2 = *(const v4f*)&qf_s[2][t][4 * c4];
        const v4f q3 = *(const v4f*)&qf_s[3][t][4 * c4];
        d0 += rr[0] * q0[0] + rr[1] * q0[1] + rr[2] * q0[2] + rr[3] * q0[3];
        d1 += rr[0] * q1[0] + rr[1] * q1[1] + rr[2] * q1[2] + rr[3] * q1[3];
        d2 += rr[0] * q2[0] + rr[1] * q2[1] + rr[2] * q2[2] + rr[3] * q2[3];
        d3 += rr[0] * q3[0] + rr[1] * q3[1] + rr[2] * q3[2] + rr[3] * q3[3];
      }
      bias_s[0][t][v] = d0;
      bias_s[1][t][v] = d1;
      bias_s[2][t][v] = d2;
      bias_s[3][t][v] = d3;
      if (it + 1 < T_ / 32)  // speculative NT prefetch of next key block
        __builtin_prefetch(rel + ((size_t)(t0 + t) * T_ + (v0 + 32 + v)) * HS_, 0, 0);
    }
    __syncthreads();

    // ---- Phase B1 (waves 0..3): scores + online softmax for 32 keys
    if (wave < 4) {
      v8f s0 = zero8(), s1 = zero8();
      s0 = WMMA_BF16(qa0, kf[0], s0);
      s0 = WMMA_BF16(qa1, kf[1], s0);
      s1 = WMMA_BF16(qa0, kf[2], s1);
      s1 = WMMA_BF16(qa1, kf[3], s1);

#pragma unroll
      for (int j = 0; j < 8; ++j) {
        const int r = j + rbase;
        // reference scales by sqrt(hs) = 8
        s0[j] = s0[j] * 8.0f + bias_s[b][r][colc];
        s1[j] = s1[j] * 8.0f + bias_s[b][r][colc + 16];

        float mc = fmaxf(s0[j], s1[j]);
#pragma unroll
        for (int msk = 1; msk < 16; msk <<= 1)
          mc = fmaxf(mc, __shfl_xor(mc, msk, 32));
        const float mnew  = fmaxf(mrow[j], mc);
        const float alpha = __expf(mrow[j] - mnew);
        const float p0 = __expf(s0[j] - mnew);
        const float p1 = __expf(s1[j] - mnew);
        float rs = p0 + p1;
#pragma unroll
        for (int msk = 1; msk < 16; msk <<= 1) rs += __shfl_xor(rs, msk, 32);
        lrow[j] = lrow[j] * alpha + rs;
        mrow[j] = mnew;
        s0[j] = p0;
        s1[j] = p1;
#pragma unroll
        for (int n = 0; n < 4; ++n) o[n][j] *= alpha;
      }
      // P tile (C layout) -> LDS row-major so it can be reloaded as an A frag
#pragma unroll
      for (int j = 0; j < 8; ++j) {
        const int r = j + rbase;
        p_s[b][r][colc]      = (bf16)s0[j];
        p_s[b][r][colc + 16] = (bf16)s1[j];
      }
    }
    __syncthreads();

    // ---- Phase B2 (waves 0..3): O += P (16x32) * V (32x64)
    if (wave < 4) {
      const v16bf pa = load_a_frag_bf16(&p_s[b][0][0], 32, lane);  // ds_load_b128
#pragma unroll
      for (int n = 0; n < 4; ++n) o[n] = WMMA_BF16(pa, vf[n], o[n]);
    }
  }

  // Epilogue: normalize by row sums, store f32 output [B,T,HS].
  if (wave < 4) {
#pragma unroll
    for (int j = 0; j < 8; ++j) {
      const float inv = 1.0f / lrow[j];
      const int   r   = t0 + j + rbase;
#pragma unroll
      for (int n = 0; n < 4; ++n)
        out[((size_t)(b * T_ + r)) * HS_ + n * 16 + colc] = o[n][j] * inv;
    }
  }
}

// ===========================================================================
extern "C" void kernel_launch(void* const* d_in, const int* in_sizes, int n_in,
                              void* d_out, int out_size, void* d_ws, size_t ws_size,
                              hipStream_t stream) {
  (void)in_sizes; (void)n_in; (void)out_size; (void)ws_size;
  const float* x   = (const float*)d_in[0];
  const float* wq  = (const float*)d_in[1];
  const float* bq  = (const float*)d_in[2];
  const float* wk  = (const float*)d_in[3];
  const float* bk  = (const float*)d_in[4];
  const float* wv  = (const float*)d_in[5];
  const float* bv  = (const float*)d_in[6];
  const float* rel = (const float*)d_in[7];
  float* out = (float*)d_out;

  // Workspace layout (~5.4 MB): q bf16, k bf16, vT bf16, q f32, wT bf16.
  bf16*  qb = (bf16*)d_ws;
  bf16*  kb = qb + (size_t)MROWS * HS_;
  bf16*  vT = kb + (size_t)MROWS * HS_;
  float* qf = (float*)(vT + (size_t)MROWS * HS_);
  bf16*  wT = (bf16*)(qf + (size_t)MROWS * HS_);

  ah_wt_convert<<<(3 * HS_ * D_ + 255) / 256, 256, 0, stream>>>(wq, wk, wv, wT);
  ah_qkv_proj<<<MROWS / 16, 96, 0, stream>>>(x, wT, bq, bk, bv, qb, kb, vT, qf);
  ah_attention<<<T_ / 16, 256, 0, stream>>>(rel, qb, kb, vT, qf, out);
}
